// ViTSelfAttention_seq_26379689132603
// MI455X (gfx1250) — matrix operand order repaired and checked by reading, not compiled, not run on hardware
//
#include <hip/hip_runtime.h>

// ---------------------------------------------------------------------------
// Types for CDNA5 WMMA (gfx1250, wave32)
// ---------------------------------------------------------------------------
typedef __attribute__((ext_vector_type(16))) __bf16 v16bf;
typedef __attribute__((ext_vector_type(8)))  float  v8f;

union BFrag { v16bf v; unsigned int u[8]; };

__device__ inline unsigned short f2bf(float f) {
  unsigned int u = __float_as_uint(f);
  u += 0x7FFFu + ((u >> 16) & 1u);          // round-to-nearest-even
  return (unsigned short)(u >> 16);
}
__device__ inline unsigned int pack2bf(float lo, float hi) {
  return (unsigned int)f2bf(lo) | ((unsigned int)f2bf(hi) << 16);
}

// ---------------------------------------------------------------------------
// Problem constants
// ---------------------------------------------------------------------------
#define HID   768
#define BATCH 8
#define SEQ   1024
#define HEADS 12
#define HDIM  64
#define ROWS  (BATCH * SEQ)          // 8192
// workspace layout (bytes)
#define XB_BYTES  ((size_t)ROWS * HID * 2)          // bf16 X
#define WB_BYTES  ((size_t)HID * HID * 2)           // bf16 W

// ---------------------------------------------------------------------------
// Kernel 1: f32 -> bf16 (2 elems per thread, dword stores)
// ---------------------------------------------------------------------------
__global__ void cvt_bf16_kernel(const float* __restrict__ src,
                                unsigned short* __restrict__ dst, int n2) {
  int i = blockIdx.x * blockDim.x + threadIdx.x;
  if (i < n2) {
    ((unsigned int*)dst)[i] = pack2bf(src[2 * i], src[2 * i + 1]);
  }
}

// ---------------------------------------------------------------------------
// Kernel 2: QKV projection GEMM.  C[8192x768] = X @ W + b, output bf16 in
// [b, head, s, d] layout.  Block = 128 threads (4 waves), tile 128x64, K=32,
// double-buffered LDS tiles.  sched_group_barrier pins a DS-read / WMMA
// software pipeline so fragment loads stay a stage ahead of the WMMAs.
// blockIdx.z selects Q/K/V; Q gets 1/sqrt(768) folded in.
// ---------------------------------------------------------------------------
__global__ __launch_bounds__(128) void qkv_gemm_kernel(
    const unsigned short* __restrict__ Xb,     // bf16 [8192][768]
    const unsigned short* __restrict__ Wall,   // bf16 3x[768][768]
    const float* __restrict__ bq, const float* __restrict__ bk,
    const float* __restrict__ bv,
    unsigned short* __restrict__ QKVb)         // bf16 3x[96][1024][64]
{
  __shared__ __align__(16) unsigned short Als[2][128][34];  // [m][k] padded
  __shared__ __align__(16) unsigned short Bls[2][64][34];   // [n][k] (W^T)

  const int tid  = threadIdx.x;
  const int w    = tid >> 5;
  const int lane = tid & 31;
  const int m0   = blockIdx.x * 128;
  const int n0   = blockIdx.y * 64;
  const int z    = blockIdx.z;

  const unsigned short* Wb = Wall + (size_t)z * HID * HID;
  const float* bias = (z == 0) ? bq : ((z == 1) ? bk : bv);
  unsigned short* outb = QKVb + (size_t)z * ROWS * HID;

  const unsigned int* Xd = (const unsigned int*)Xb;
  const unsigned int* Wd = (const unsigned int*)Wb;

  v8f acc[2][4] = {};

  auto stage = [&](int buf, int kt2) {
    const int k0 = kt2 * 32;
    // A tile: 128 rows x 16 dwords
#pragma unroll
    for (int i = 0; i < 16; ++i) {
      int lin = tid + 128 * i;                 // 0..2047
      int r = lin >> 4, c = lin & 15;
      unsigned int d = Xd[(size_t)(m0 + r) * (HID / 2) + (k0 >> 1) + c];
      *(unsigned int*)&Als[buf][r][2 * c] = d;
    }
    // B tile transposed: W[k0..+31][n0..+63] -> Bls[n][k]
#pragma unroll
    for (int i = 0; i < 8; ++i) {
      int lin = tid + 128 * i;                 // 0..1023
      int kk = lin >> 5, np = lin & 31;
      unsigned int d = Wd[(size_t)(k0 + kk) * (HID / 2) + (n0 >> 1) + np];
      Bls[buf][2 * np][kk]     = (unsigned short)(d & 0xFFFFu);
      Bls[buf][2 * np + 1][kk] = (unsigned short)(d >> 16);
    }
    // prefetch the tile after this one (global_prefetch_b8)
    if (kt2 + 1 < HID / 32) {
      const int kn = k0 + 32;
      __builtin_prefetch(&Xd[(size_t)(m0 + (tid >> 4) * 8) * (HID / 2) +
                             (kn >> 1) + (tid & 15)], 0, 1);
      __builtin_prefetch(&Wd[(size_t)(kn + (tid >> 5) * 4) * (HID / 2) +
                             (n0 >> 1) + (tid & 31)], 0, 1);
    }
  };

  // B-fragment loader
  auto loadB = [&](BFrag& dst, int t, int buf) {
    int n  = t * 16 + (lane & 15);
    int kh = (lane >> 4) * 8;
#pragma unroll
    for (int j = 0; j < 8; ++j)
      dst.u[j] = *(const unsigned int*)&Bls[buf][n][kh + (j >> 2) * 16 + 2 * (j & 3)];
  };

  stage(0, 0);

  for (int kt = 0; kt < HID / 32; ++kt) {
    const int cur = kt & 1;
    __syncthreads();                        // staged tile `cur` is ready
    if (kt + 1 < HID / 32) stage(cur ^ 1, kt + 1);

    // two A fragments: rows w*32 .. w*32+31
    BFrag a0, a1;
    {
      int m   = w * 32 + (lane & 15);
      int kb0 = (lane >> 4) * 8;
#pragma unroll
      for (int j = 0; j < 8; ++j) {
        int k = ((j >> 2) * 16) + kb0 + 2 * (j & 3);
        a0.u[j] = *(const unsigned int*)&Als[cur][m][k];
        a1.u[j] = *(const unsigned int*)&Als[cur][m + 16][k];
      }
    }
    // preload all four B-fragments
    BFrag b0, b1, b2, b3;
    loadB(b0, 0, cur);
    loadB(b1, 1, cur);
    loadB(b2, 2, cur);
    loadB(b3, 3, cur);
    // WMMA burst
    acc[0][0] = __builtin_amdgcn_wmma_f32_16x16x32_bf16(
        false, a0.v, false, b0.v, (short)0, acc[0][0], false, false);
    acc[1][0] = __builtin_amdgcn_wmma_f32_16x16x32_bf16(
        false, a1.v, false, b0.v, (short)0, acc[1][0], false, false);
    acc[0][1] = __builtin_amdgcn_wmma_f32_16x16x32_bf16(
        false, a0.v, false, b1.v, (short)0, acc[0][1], false, false);
    acc[1][1] = __builtin_amdgcn_wmma_f32_16x16x32_bf16(
        false, a1.v, false, b1.v, (short)0, acc[1][1], false, false);
    acc[0][2] = __builtin_amdgcn_wmma_f32_16x16x32_bf16(
        false, a0.v, false, b2.v, (short)0, acc[0][2], false, false);
    acc[1][2] = __builtin_amdgcn_wmma_f32_16x16x32_bf16(
        false, a1.v, false, b2.v, (short)0, acc[1][2], false, false);
    acc[0][3] = __builtin_amdgcn_wmma_f32_16x16x32_bf16(
        false, a0.v, false, b3.v, (short)0, acc[0][3], false, false);
    acc[1][3] = __builtin_amdgcn_wmma_f32_16x16x32_bf16(
        false, a1.v, false, b3.v, (short)0, acc[1][3], false, false);

    // Pin a DS-read / WMMA pipeline for this iteration:
    //   16 DS reads (a0,a1,b0,b1) -> 2 WMMA (b0) -> 4 DS (b2) -> 2 WMMA (b1)
    //   -> 4 DS (b3) -> 4 WMMA (b2,b3)
    // mask 0x100 = DS read, 0x8 = MFMA/WMMA group.
    __builtin_amdgcn_sched_group_barrier(0x100, 16, 0);
    __builtin_amdgcn_sched_group_barrier(0x008, 2, 0);
    __builtin_amdgcn_sched_group_barrier(0x100, 4, 0);
    __builtin_amdgcn_sched_group_barrier(0x008, 2, 0);
    __builtin_amdgcn_sched_group_barrier(0x100, 4, 0);
    __builtin_amdgcn_sched_group_barrier(0x008, 4, 0);
  }

  // epilogue: bias, optional Q scale, store bf16 in [b,h,s,d]
  const float qscale = 0.03608439182435161f;   // 1/sqrt(768)
  const int hl = lane >> 4, nn = lane & 15;
#pragma unroll
  for (int mf = 0; mf < 2; ++mf) {
#pragma unroll
    for (int t = 0; t < 4; ++t) {
      int n = n0 + t * 16 + nn;
      float bval = bias[n];
      int hh = n >> 6, dd = n & 63;
#pragma unroll
      for (int r = 0; r < 8; ++r) {
        int m = m0 + w * 32 + mf * 16 + r + 8 * hl;
        int bI = m >> 10, s = m & 1023;
        float val = acc[mf][t][r] + bval;
        if (z == 0) val *= qscale;
        outb[(((size_t)(bI * HEADS + hh)) << 16) + ((size_t)s << 6) + dd] =
            f2bf(val);
      }
    }
  }
}

// ---------------------------------------------------------------------------
// Kernel 3: flash attention.  grid = (16 q-tiles, 96 b*h), block = 128 (4
// waves).  Each wave owns 16 query rows, streams 32-key tiles (double-
// buffered K/V in LDS), keeps the 16x64 output in WMMA accumulators with
// online softmax.
// ---------------------------------------------------------------------------
__global__ __launch_bounds__(128) void attn_kernel(
    const unsigned short* __restrict__ Qb,   // bf16 [96][1024][64] (pre-scaled)
    const unsigned short* __restrict__ Kb,
    const unsigned short* __restrict__ Vb,
    float* __restrict__ out)                 // f32 [8][1024][768]
{
  __shared__ __align__(16) unsigned short Kls[2][32][66];   // [key][d]
  __shared__ __align__(16) unsigned short Vls[2][64][34];   // [d][key] (V^T)
  __shared__ __align__(16) unsigned short Pls[4][16][34];   // per-wave P tile

  const int tid = threadIdx.x, w = tid >> 5, lane = tid & 31;
  const int qt = blockIdx.x;               // 0..15
  const int bh = blockIdx.y;               // 0..95
  const int b = bh / HEADS, h = bh % HEADS;
  const size_t base = (size_t)bh << 16;    // elem offset into [bh][1024][64]
  const int q0 = qt * 64 + w * 16;

  // Q fragments (hd 0..31 and 32..63), loaded straight in A-layout
  BFrag qf0, qf1;
  {
    const unsigned int* Qd = (const unsigned int*)(Qb + base);
    int m   = q0 + (lane & 15);
    int kb0 = (lane >> 4) * 8;
#pragma unroll
    for (int j = 0; j < 8; ++j) {
      int d = ((j >> 2) * 16) + kb0 + 2 * (j & 3);
      qf0.u[j] = Qd[(size_t)m * 32 + (d >> 1)];
      qf1.u[j] = Qd[(size_t)m * 32 + ((d + 32) >> 1)];
    }
  }

  v8f O[4] = {};
  float mrow[8], lrow[8];
#pragma unroll
  for (int r = 0; r < 8; ++r) { mrow[r] = -3.0e38f; lrow[r] = 0.0f; }

  const unsigned int* Kd = (const unsigned int*)(Kb + base);
  const unsigned int* Vd = (const unsigned int*)(Vb + base);

  auto stage = [&](int buf, int kt2) {
    // 32 keys of K (row-major) and V (transposed)
#pragma unroll
    for (int i = 0; i < 8; ++i) {
      int lin = tid + 128 * i;               // 0..1023 dwords
      int key = lin >> 5, dp = lin & 31;
      unsigned int kd = Kd[(size_t)(kt2 * 32 + key) * 32 + dp];
      *(unsigned int*)&Kls[buf][key][2 * dp] = kd;
      unsigned int vd = Vd[(size_t)(kt2 * 32 + key) * 32 + dp];
      Vls[buf][2 * dp][key]     = (unsigned short)(vd & 0xFFFFu);
      Vls[buf][2 * dp + 1][key] = (unsigned short)(vd >> 16);
    }
    if (kt2 + 1 < SEQ / 32) {
      __builtin_prefetch(&Kd[(size_t)(kt2 + 1) * 1024 + tid * 8], 0, 1);
      __builtin_prefetch(&Vd[(size_t)(kt2 + 1) * 1024 + tid * 8], 0, 1);
    }
  };

  // K-fragment loader: B-matrix for score WMMA (tile = key group, hc = hd32)
  auto loadK = [&](BFrag& dst, int hc, int tile, int buf) {
    int kh   = hc * 32 + (lane >> 4) * 8;
    int key0 = (lane & 15) + tile * 16;
#pragma unroll
    for (int j = 0; j < 8; ++j)
      dst.u[j] = *(const unsigned int*)&Kls[buf][key0][kh + (j >> 2) * 16 + 2 * (j & 3)];
  };
  // V-fragment loader: B-matrix for PV WMMA (tile = d group)
  auto loadV = [&](BFrag& dst, int t, int buf) {
    int dloc = t * 16 + (lane & 15);
    int kh   = (lane >> 4) * 8;
#pragma unroll
    for (int j = 0; j < 8; ++j)
      dst.u[j] = *(const unsigned int*)&Vls[buf][dloc][kh + (j >> 2) * 16 + 2 * (j & 3)];
  };

  stage(0, 0);

  for (int kt = 0; kt < SEQ / 32; ++kt) {
    const int cur = kt & 1;
    __syncthreads();                        // staged tile `cur` is ready
    if (kt + 1 < SEQ / 32) stage(cur ^ 1, kt + 1);

    // scores: preload all four K-frags, then 4 WMMAs back-to-back
    v8f s0 = {}, s1 = {};
    {
      BFrag k00, k01, k10, k11;
      loadK(k00, 0, 0, cur);
      loadK(k01, 0, 1, cur);
      loadK(k10, 1, 0, cur);
      loadK(k11, 1, 1, cur);
      s0 = __builtin_amdgcn_wmma_f32_16x16x32_bf16(
          false, qf0.v, false, k00.v, (short)0, s0, false, false);
      s1 = __builtin_amdgcn_wmma_f32_16x16x32_bf16(
          false, qf0.v, false, k01.v, (short)0, s1, false, false);
      s0 = __builtin_amdgcn_wmma_f32_16x16x32_bf16(
          false, qf1.v, false, k10.v, (short)0, s0, false, false);
      s1 = __builtin_amdgcn_wmma_f32_16x16x32_bf16(
          false, qf1.v, false, k11.v, (short)0, s1, false, false);
    }

    // online softmax; row (r + 8*lane_half) is spread across 16 lanes
    const int hl = lane >> 4, nn = lane & 15;
#pragma unroll
    for (int r = 0; r < 8; ++r) {
      float tm = fmaxf(s0[r], s1[r]);
#pragma unroll
      for (int off = 1; off < 16; off <<= 1)
        tm = fmaxf(tm, __shfl_xor(tm, off, 16));
      float mnew = fmaxf(mrow[r], tm);
      float cf = __expf(mrow[r] - mnew);
      float p0 = __expf(s0[r] - mnew);
      float p1 = __expf(s1[r] - mnew);
      float ps = p0 + p1;
#pragma unroll
      for (int off = 1; off < 16; off <<= 1)
        ps += __shfl_xor(ps, off, 16);
      lrow[r] = lrow[r] * cf + ps;
      mrow[r] = mnew;
      O[0][r] *= cf; O[1][r] *= cf; O[2][r] *= cf; O[3][r] *= cf;
      int mloc = r + 8 * hl;
      Pls[w][mloc][nn]      = f2bf(p0);
      Pls[w][mloc][nn + 16] = f2bf(p1);
    }
    asm volatile("s_wait_dscnt 0" ::: "memory");   // wave-local LDS RAW

    // P as A-fragment (16 q x 32 keys)
    BFrag pf;
    {
      int m   = lane & 15;
      int kb0 = (lane >> 4) * 8;
#pragma unroll
      for (int j = 0; j < 8; ++j) {
        int k = ((j >> 2) * 16) + kb0 + 2 * (j & 3);
        pf.u[j] = *(const unsigned int*)&Pls[w][m][k];
      }
    }
    // O += P @ V: preload all four V-frags, then 4 WMMAs back-to-back
    {
      BFrag v0, v1, v2, v3;
      loadV(v0, 0, cur);
      loadV(v1, 1, cur);
      loadV(v2, 2, cur);
      loadV(v3, 3, cur);
      O[0] = __builtin_amdgcn_wmma_f32_16x16x32_bf16(
          false, pf.v, false, v0.v, (short)0, O[0], false, false);
      O[1] = __builtin_amdgcn_wmma_f32_16x16x32_bf16(
          false, pf.v, false, v1.v, (short)0, O[1], false, false);
      O[2] = __builtin_amdgcn_wmma_f32_16x16x32_bf16(
          false, pf.v, false, v2.v, (short)0, O[2], false, false);
      O[3] = __builtin_amdgcn_wmma_f32_16x16x32_bf16(
          false, pf.v, false, v3.v, (short)0, O[3], false, false);
    }
  }

  // normalize and write [B,S,H] f32
  const int hl = lane >> 4, nn = lane & 15;
#pragma unroll
  for (int t = 0; t < 4; ++t) {
#pragma unroll
    for (int r = 0; r < 8; ++r) {
      int q = q0 + r + 8 * hl;
      int d = t * 16 + nn;
      out[((size_t)(b * SEQ + q)) * HID + h * HDIM + d] = O[t][r] / lrow[r];
    }
  }
}

// ---------------------------------------------------------------------------
// Host launcher
// ---------------------------------------------------------------------------
extern "C" void kernel_launch(void* const* d_in, const int* in_sizes, int n_in,
                              void* d_out, int out_size, void* d_ws,
                              size_t ws_size, hipStream_t stream) {
  const float* hidden = (const float*)d_in[0];
  const float* Wq = (const float*)d_in[1];
  const float* bq = (const float*)d_in[2];
  const float* Wk = (const float*)d_in[3];
  const float* bk = (const float*)d_in[4];
  const float* Wv = (const float*)d_in[5];
  const float* bv = (const float*)d_in[6];
  float* out = (float*)d_out;

  char* ws = (char*)d_ws;
  unsigned short* Xb   = (unsigned short*)ws;
  unsigned short* Wall = (unsigned short*)(ws + XB_BYTES);
  unsigned short* QKVb = (unsigned short*)(ws + XB_BYTES + 3 * WB_BYTES);

  // f32 -> bf16 conversions
  {
    int nX2 = ROWS * HID / 2;
    cvt_bf16_kernel<<<(nX2 + 255) / 256, 256, 0, stream>>>(hidden, Xb, nX2);
    int nW2 = HID * HID / 2;
    cvt_bf16_kernel<<<(nW2 + 255) / 256, 256, 0, stream>>>(Wq, Wall, nW2);
    cvt_bf16_kernel<<<(nW2 + 255) / 256, 256, 0, stream>>>(
        Wk, Wall + (size_t)HID * HID, nW2);
    cvt_bf16_kernel<<<(nW2 + 255) / 256, 256, 0, stream>>>(
        Wv, Wall + (size_t)2 * HID * HID, nW2);
  }

  // QKV projection GEMMs (WMMA bf16)
  qkv_gemm_kernel<<<dim3(ROWS / 128, HID / 64, 3), 128, 0, stream>>>(
      Xb, Wall, bq, bk, bv, QKVb);

  // flash attention (WMMA bf16, online softmax)
  attn_kernel<<<dim3(SEQ / 64, BATCH * HEADS), 128, 0, stream>>>(
      QKVb, QKVb + (size_t)ROWS * HID, QKVb + (size_t)2 * ROWS * HID, out);
}
